// RNNDecoder_44100724195738
// MI455X (gfx1250) — matrix operand order repaired and checked by reading, not compiled, not run on hardware
//
#include <hip/hip_runtime.h>
#include <hip/hip_bf16.h>
#include <math.h>

// ---------------- problem constants ----------------
#define B_    64
#define T_    1024
#define U_    512
#define ENC_  128
#define H_    512
#define L_    3
#define KATT  10
#define OUTD  121            // 6*20+1
#define G4H   2048           // 4*H
#define KP0   672            // 3+ENC+H = 643 padded to mult of 32
#define KPN   1184           // 3+ENC+H+H = 1155 padded to mult of 32
#define KT0   (KP0/32)       // 21
#define KTN   (KPN/32)       // 37

typedef __attribute__((ext_vector_type(16))) __bf16 v16bf;
typedef __attribute__((ext_vector_type(8)))  __bf16 v8bf;
typedef __attribute__((ext_vector_type(8)))  float  v8f;
typedef unsigned int u32x4 __attribute__((ext_vector_type(4)));
typedef int          i32x8 __attribute__((ext_vector_type(8)));
typedef int          i32x4 __attribute__((ext_vector_type(4)));

#if defined(__has_builtin)
#if __has_builtin(__builtin_amdgcn_tensor_load_to_lds) && \
    __has_builtin(__builtin_amdgcn_s_wait_tensorcnt)
#define USE_TDM 1
#else
#define USE_TDM 0
#endif
#else
#define USE_TDM 0
#endif

__device__ __forceinline__ float sigf(float x) { return 1.0f / (1.0f + expf(-x)); }

__device__ __forceinline__ v16bf load_bf16_frag(const __bf16* p) {
    v8bf lo = *(const v8bf*)p;
    v8bf hi = *(const v8bf*)(p + 16);
    return __builtin_shufflevector(lo, hi, 0,1,2,3,4,5,6,7,8,9,10,11,12,13,14,15);
}

#if USE_TDM
// TDM: async-copy one wave's B slice for K-tile kt into LDS.
// Slice = 2 rows (sel=0/1) x 512B (16 cols x 16 bf16), row stride 64KB in the
// packed weight array -> 2D tensor descriptor, data_size=2B, dims in elements.
// This toolchain's builtin arity is 6: (g0, g1, g2, g3, g4, cpol).
__device__ __forceinline__ void tdm_load_b(const void* gaddr, unsigned lds_off) {
    unsigned long long ga = (unsigned long long)(uintptr_t)gaddr;
    u32x4 g0;
    g0[0] = 1u;                                                  // count=1 (valid D#)
    g0[1] = lds_off;                                             // LDS byte address
    g0[2] = (unsigned)ga;                                        // global_addr[31:0]
    g0[3] = (unsigned)((ga >> 32) & 0x01FFFFFFu) | (2u << 30);   // [120:64] + type=2
    i32x8 g1;
    g1[0] = (int)(1u << 16);      // workgroup_mask=0, data_size=1 (2 bytes)
    g1[1] = (int)(256u << 16);    // tensor_dim0[15:0]=256 elems (bits 63:48)
    g1[2] = (int)(2u << 16);      // tensor_dim0 hi=0 | tensor_dim1[15:0]=2
    g1[3] = (int)(256u << 16);    // tensor_dim1 hi=0 | tile_dim0=256
    g1[4] = 2;                    // tile_dim1=2, tile_dim2=0
    g1[5] = 32768;                // tensor_dim0_stride[31:0] = 2048*16 elems (64KB)
    g1[6] = 0;                    // stride0 hi | tensor_dim1_stride lo
    g1[7] = 0;
    i32x4 z4 = {0, 0, 0, 0};              // groups 2/3 unused (<=2D tensor)
    i32x8 z8 = {0, 0, 0, 0, 0, 0, 0, 0};  // extra descriptor group (unused)
    __builtin_amdgcn_tensor_load_to_lds(g0, g1, z4, z4, z8, 0);
}
#endif

// ---------------- weight packing (once per call) ----------------
// Packed B layout (bf16): Wp[kt][sel][n][i], i in 0..15 contiguous.
// Lane (sel = lane>>4, n = n0 + (lane&15)) reads 16 contiguous bf16 = the 16
// K-values (k = kt*32 + (i>>3)*16 + sel*8 + (i&7)) of the 32x16 bf16 B fragment.
__global__ void pack_w0_k(const float* __restrict__ Wih, const float* __restrict__ Whh,
                          __bf16* __restrict__ Wp) {
    int idx = blockIdx.x * blockDim.x + threadIdx.x;
    if (idx >= KT0 * 2 * G4H * 16) return;
    int i   = idx & 15;
    int n   = (idx >> 4) & (G4H - 1);
    int sel = (idx >> 15) & 1;
    int kt  = idx >> 16;
    int k   = (kt << 5) + ((i >> 3) << 4) + (sel << 3) + (i & 7);
    float v = 0.0f;
    if (k < 131)      v = Wih[(size_t)n * 131 + k];          // [x_t, w] block
    else if (k < 643) v = Whh[(size_t)n * 512 + (k - 131)];  // recurrent block
    Wp[idx] = (__bf16)v;
}

__global__ void pack_wn_k(const float* __restrict__ Wihn, const float* __restrict__ Whhn,
                          __bf16* __restrict__ Wp, int j) {
    int idx = blockIdx.x * blockDim.x + threadIdx.x;
    if (idx >= KTN * 2 * G4H * 16) return;
    int i   = idx & 15;
    int n   = (idx >> 4) & (G4H - 1);
    int sel = (idx >> 15) & 1;
    int kt  = idx >> 16;
    int k   = (kt << 5) + ((i >> 3) << 4) + (sel << 3) + (i & 7);
    float v = 0.0f;
    if (k < 643)       v = Wihn[(size_t)j * G4H * 643 + (size_t)n * 643 + k];
    else if (k < 1155) v = Whhn[(size_t)j * G4H * 512 + (size_t)n * 512 + (k - 643)];
    Wp[idx] = (__bf16)v;
}

// ---------------- state init ----------------
__global__ void init_k(const float* __restrict__ h0, const float* __restrict__ c0,
                       const float* __restrict__ w0, const float* __restrict__ k0,
                       float* __restrict__ h_s, float* __restrict__ c_s,
                       float* __restrict__ w_s, float* __restrict__ k_s) {
    int idx = blockIdx.x * blockDim.x + threadIdx.x;
    if (idx < L_ * B_ * H_) {
        int l = idx / (B_ * H_);
        int u = idx & (H_ - 1);
        h_s[idx] = h0[l * H_ + u];
        c_s[idx] = c0[l * H_ + u];
    }
    if (idx < B_ * ENC_) w_s[idx] = w0[idx & (ENC_ - 1)];
    if (idx < B_ * KATT) k_s[idx] = k0[idx % KATT];
}

// ---------------- per-step activation build (f32 state -> bf16 A matrix) ----------------
__global__ void build_a0_k(const float* __restrict__ strokes, const float* __restrict__ w_s,
                           const float* __restrict__ h_s, __bf16* __restrict__ A, int t) {
    int idx = blockIdx.x * blockDim.x + threadIdx.x;
    if (idx >= B_ * KP0) return;
    int b = idx / KP0, k = idx - b * KP0;
    float v = 0.0f;
    if (k < 3)                  v = strokes[((size_t)b * T_ + t) * 3 + k];
    else if (k < 3 + ENC_)      v = w_s[b * ENC_ + (k - 3)];
    else if (k < 3 + ENC_ + H_) v = h_s[(size_t)b * H_ + (k - 3 - ENC_)];
    A[idx] = (__bf16)v;
}

// layer j in {1,2}: input = [x_t, w_new, h_{j-1}(new)] ++ recurrent h_j(prev)
__global__ void build_an_k(const float* __restrict__ strokes, const float* __restrict__ w_s,
                           const float* __restrict__ h_s, __bf16* __restrict__ A, int t, int j) {
    int idx = blockIdx.x * blockDim.x + threadIdx.x;
    if (idx >= B_ * KPN) return;
    int b = idx / KPN, k = idx - b * KPN;
    float v = 0.0f;
    if (k < 3)         v = strokes[((size_t)b * T_ + t) * 3 + k];
    else if (k < 131)  v = w_s[b * ENC_ + (k - 3)];
    else if (k < 643)  v = h_s[(size_t)(j - 1) * B_ * H_ + (size_t)b * H_ + (k - 131)];
    else if (k < 1155) v = h_s[(size_t)j * B_ * H_ + (size_t)b * H_ + (k - 643)];
    A[idx] = (__bf16)v;
}

// ---------------- bf16 WMMA GEMM: gates[64,2048] = A[64,Kpad] x Bp ----------------
// One wave owns one 16-wide N column and ALL of M=64 (4 independent C tiles ->
// 4-way WMMA ILP, B fragment reused 4x). 4 waves/block, grid.x=32 covers N=2048.
// B panel is TDM double-buffered into a per-wave LDS slice (no cross-wave sync:
// TENSORcnt is per-wave and in-order); A is read directly (small, cache-hot).
__global__ __launch_bounds__(128) void wmma_gemm_k(const __bf16* __restrict__ A,
                                                   const __bf16* __restrict__ Bp,
                                                   float* __restrict__ out, int Kpad) {
#if USE_TDM
    __shared__ __align__(128) unsigned char smem[4][2][1024];
#endif
    const int lane  = threadIdx.x & 31;
    const int wv    = threadIdx.x >> 5;           // 0..3 -> n-tile within block
    const int ntile = (blockIdx.x << 2) + wv;     // 0..127
    const int n0    = ntile << 4;
    const int sel   = lane >> 4;                  // lane half selects K-subgroups
    const int l15   = lane & 15;
    const int ktn   = Kpad >> 5;

    v8f c0 = {}, c1 = {}, c2 = {}, c3 = {};

#if USE_TDM
    const unsigned char* gB = (const unsigned char*)Bp;
    unsigned ldsbase = (unsigned)(uintptr_t)&smem[wv][0][0];
    tdm_load_b(gB + (size_t)n0 * 32, ldsbase);            // prefetch kt=0 -> buf0
    int cur = 0;
#endif
    for (int kt = 0; kt < ktn; ++kt) {
#if USE_TDM
        if (kt + 1 < ktn) {
            // prefetch next K-tile into the other buffer, then wait for current
            tdm_load_b(gB + (size_t)(kt + 1) * 131072 + (size_t)n0 * 32,
                       ldsbase + (unsigned)((cur ^ 1) << 10));
            __builtin_amdgcn_s_wait_tensorcnt(1);
        } else {
            __builtin_amdgcn_s_wait_tensorcnt(0);
        }
        const unsigned char* bb = &smem[wv][cur][0] + (sel << 9) + (l15 << 5);
        v8bf blo = *(const v8bf*)(bb);
        v8bf bhi = *(const v8bf*)(bb + 16);
        v16bf b = __builtin_shufflevector(blo, bhi, 0,1,2,3,4,5,6,7,8,9,10,11,12,13,14,15);
        cur ^= 1;
#else
        const __bf16* bptr = Bp + ((((size_t)(kt * 2 + sel)) << 11) + (size_t)(n0 + l15)) * 16;
        v8bf blo = *(const v8bf*)bptr;
        v8bf bhi = *(const v8bf*)(bptr + 8);
        v16bf b = __builtin_shufflevector(blo, bhi, 0,1,2,3,4,5,6,7,8,9,10,11,12,13,14,15);  // 16 contiguous bf16
#endif
        // A fragments (16x32 bf16) for the 4 M tiles; lane half sel picks K
        // groups {kt*32+sel*8..} and {kt*32+16+sel*8..} per the ISA layout.
        const __bf16* ar = A + (size_t)l15 * Kpad + (kt << 5) + (sel << 3);
        v16bf a0 = load_bf16_frag(ar);
        v16bf a1 = load_bf16_frag(ar + (size_t)16 * Kpad);
        v16bf a2 = load_bf16_frag(ar + (size_t)32 * Kpad);
        v16bf a3 = load_bf16_frag(ar + (size_t)48 * Kpad);
        c0 = __builtin_amdgcn_wmma_f32_16x16x32_bf16(false, a0, false, b, (short)0, c0, false, false);
        c1 = __builtin_amdgcn_wmma_f32_16x16x32_bf16(false, a1, false, b, (short)0, c1, false, false);
        c2 = __builtin_amdgcn_wmma_f32_16x16x32_bf16(false, a2, false, b, (short)0, c2, false, false);
        c3 = __builtin_amdgcn_wmma_f32_16x16x32_bf16(false, a3, false, b, (short)0, c3, false, false);
    }
    // C layout: VGPR r: lanes 0-15 -> M=r, lanes 16-31 -> M=r+8; N = n0+(lane&15)
    float* orow = out + (n0 + l15);
#pragma unroll
    for (int r = 0; r < 8; ++r) {
        int m = (sel << 3) + r;
        orow[(size_t)(m)      * G4H] = c0[r];
        orow[(size_t)(m + 16) * G4H] = c1[r];
        orow[(size_t)(m + 32) * G4H] = c2[r];
        orow[(size_t)(m + 48) * G4H] = c3[r];
    }
}

// ---------------- LSTM cell elementwise (adds biases, updates h/c in place) -------------
__global__ __launch_bounds__(256) void lstm_cell_k(const float* __restrict__ gates,
                                                   const float* __restrict__ b_ih,
                                                   const float* __restrict__ b_hh,
                                                   float* __restrict__ h_st,
                                                   float* __restrict__ c_st) {
    int idx = blockIdx.x * blockDim.x + threadIdx.x;
    if (idx >= B_ * H_) return;
    int b = idx >> 9, n = idx & (H_ - 1);
    const float* g = gates + (size_t)b * G4H;
    float ig = g[n]          + b_ih[n]          + b_hh[n];
    float fg = g[H_ + n]     + b_ih[H_ + n]     + b_hh[H_ + n];
    float gg = g[2 * H_ + n] + b_ih[2 * H_ + n] + b_hh[2 * H_ + n];
    float og = g[3 * H_ + n] + b_ih[3 * H_ + n] + b_hh[3 * H_ + n];
    float cn = sigf(fg) * c_st[idx] + sigf(ig) * tanhf(gg);
    float hn = sigf(og) * tanhf(cn);
    c_st[idx] = cn;
    h_st[idx] = hn;
}

// ---------------- Gaussian mixture attention (one block per batch row) ----------------
__global__ __launch_bounds__(512) void attention_k(const float* __restrict__ h0,
                                                   const float* __restrict__ Watt,
                                                   const float* __restrict__ batt,
                                                   const float* __restrict__ context,
                                                   float* __restrict__ k_state,
                                                   float* __restrict__ w_state,
                                                   float* __restrict__ att_out, int t) {
    __shared__ float alpha[KATT], beta[KATT], kapp[KATT];
    __shared__ float phi_s[U_];
    const int b = blockIdx.x;
    const int tid = threadIdx.x;
    if (tid < 3 * KATT) {
        const float* hrow = h0 + (size_t)b * H_;
        const float* wrow = Watt + (size_t)tid * H_;
        float acc = batt[tid];
        for (int i = 0; i < H_; ++i) acc += hrow[i] * wrow[i];
        float e = expf(acc);
        int k = tid % KATT, grp = tid / KATT;
        if (grp == 0) alpha[k] = e;
        else if (grp == 1) beta[k] = e;
        else {
            float kn = e + k_state[b * KATT + k];  // kappa = exp(raw) + k_prev
            kapp[k] = kn;
            k_state[b * KATT + k] = kn;
        }
    }
    __syncthreads();
    {   // phi[u] = sum_k alpha_k * exp(-beta_k * (kappa_k - u)^2)
        float u = (float)tid;
        float p = 0.0f;
#pragma unroll
        for (int k = 0; k < KATT; ++k) {
            float d = kapp[k] - u;
            p += alpha[k] * expf(-beta[k] * d * d);
        }
        phi_s[tid] = p;
        att_out[((size_t)b * T_ + t) * U_ + tid] = p;
    }
    __syncthreads();
    if (tid < ENC_) {  // w_new[e] = sum_u phi[u] * context[b,u,e]
        const float* ctx = context + (size_t)b * U_ * ENC_ + tid;
        float acc = 0.0f;
        for (int u = 0; u < U_; ++u) acc += phi_s[u] * ctx[(size_t)u * ENC_];
        w_state[b * ENC_ + tid] = acc;
    }
}

// ---------------- output projection: out[b,t,:] = [h0|h1|h2] @ W_out^T + b_out ---------
__global__ __launch_bounds__(128) void out_k(const float* __restrict__ h_s,
                                             const float* __restrict__ Wout,
                                             const float* __restrict__ bout,
                                             float* __restrict__ out, int t) {
    int b = blockIdx.x, o = threadIdx.x;
    if (o >= OUTD) return;
    float acc = bout[o];
    const float* wrow = Wout + (size_t)o * (L_ * H_);
    for (int l = 0; l < L_; ++l) {
        const float* hr = h_s + (size_t)l * B_ * H_ + (size_t)b * H_;
        const float* wr = wrow + l * H_;
        for (int i = 0; i < H_; ++i) acc += hr[i] * wr[i];
    }
    out[((size_t)b * T_ + t) * OUTD + o] = acc;
}

// ---------------- host ----------------
extern "C" void kernel_launch(void* const* d_in, const int* in_sizes, int n_in,
                              void* d_out, int out_size, void* d_ws, size_t ws_size,
                              hipStream_t stream) {
    (void)in_sizes; (void)n_in; (void)out_size; (void)ws_size;
    const float* strokes = (const float*)d_in[0];
    const float* context = (const float*)d_in[1];
    const float* W_ih0   = (const float*)d_in[2];
    const float* W_hh0   = (const float*)d_in[3];
    const float* b_ih0   = (const float*)d_in[4];
    const float* b_hh0   = (const float*)d_in[5];
    const float* W_ihn   = (const float*)d_in[6];
    const float* W_hhn   = (const float*)d_in[7];
    const float* b_ihn   = (const float*)d_in[8];
    const float* b_hhn   = (const float*)d_in[9];
    const float* W_att   = (const float*)d_in[10];
    const float* b_att   = (const float*)d_in[11];
    const float* W_out   = (const float*)d_in[12];
    const float* b_out   = (const float*)d_in[13];
    const float* h_0     = (const float*)d_in[14];
    const float* c_0     = (const float*)d_in[15];
    const float* w_0     = (const float*)d_in[16];
    const float* k_0     = (const float*)d_in[17];

    // ---- workspace layout ----
    float* ws_f  = (float*)d_ws;
    float* h_s   = ws_f;                       // [L,B,H]  = h_f layout
    float* c_s   = h_s + L_ * B_ * H_;         // [L,B,H]
    float* w_s   = c_s + L_ * B_ * H_;         // [B,ENC]
    float* k_s   = w_s + B_ * ENC_;            // [B,KATT]
    float* gates = k_s + B_ * KATT;            // [B,4H]
    __bf16* A0   = (__bf16*)(gates + B_ * G4H);
    __bf16* A1   = A0 + (size_t)B_ * KP0;
    __bf16* Wp0  = A1 + (size_t)B_ * KPN;
    __bf16* WpN0 = Wp0 + (size_t)KT0 * 2 * G4H * 16;
    __bf16* WpN1 = WpN0 + (size_t)KTN * 2 * G4H * 16;

    // ---- output layout: outputs | attention | h_f | c_f | w_f | k_f ----
    float* out_f    = (float*)d_out;
    float* out_main = out_f;
    float* out_att  = out_main + (size_t)B_ * T_ * OUTD;
    float* out_h    = out_att + (size_t)B_ * T_ * U_;
    float* out_c    = out_h + (size_t)L_ * B_ * H_;
    float* out_w    = out_c + (size_t)L_ * B_ * H_;
    float* out_kf   = out_w + (size_t)B_ * ENC_;

    // ---- pack weights to bf16 fragment order + init states (every call) ----
    {
        int n0 = KT0 * 2 * G4H * 16;
        pack_w0_k<<<(n0 + 255) / 256, 256, 0, stream>>>(W_ih0, W_hh0, Wp0);
        int nn = KTN * 2 * G4H * 16;
        pack_wn_k<<<(nn + 255) / 256, 256, 0, stream>>>(W_ihn, W_hhn, WpN0, 0);
        pack_wn_k<<<(nn + 255) / 256, 256, 0, stream>>>(W_ihn, W_hhn, WpN1, 1);
        int ni = L_ * B_ * H_;
        init_k<<<(ni + 255) / 256, 256, 0, stream>>>(h_0, c_0, w_0, k_0, h_s, c_s, w_s, k_s);
    }

    // ---- recurrent loop ----
    for (int t = 0; t < T_; ++t) {
        // layer 0: input [x_t, w_prev] ++ recurrent h0_prev
        build_a0_k<<<(B_ * KP0 + 255) / 256, 256, 0, stream>>>(strokes, w_s, h_s, A0, t);
        wmma_gemm_k<<<32, 128, 0, stream>>>(A0, Wp0, gates, KP0);
        lstm_cell_k<<<(B_ * H_ + 255) / 256, 256, 0, stream>>>(gates, b_ih0, b_hh0, h_s, c_s);

        // attention: consumes h0_new, produces w_new + kappa + phi
        attention_k<<<B_, 512, 0, stream>>>(h_s, W_att, b_att, context, k_s, w_s, out_att, t);

        // layer 1
        build_an_k<<<(B_ * KPN + 255) / 256, 256, 0, stream>>>(strokes, w_s, h_s, A1, t, 1);
        wmma_gemm_k<<<32, 128, 0, stream>>>(A1, WpN0, gates, KPN);
        lstm_cell_k<<<(B_ * H_ + 255) / 256, 256, 0, stream>>>(
            gates, b_ihn, b_hhn, h_s + (size_t)B_ * H_, c_s + (size_t)B_ * H_);

        // layer 2
        build_an_k<<<(B_ * KPN + 255) / 256, 256, 0, stream>>>(strokes, w_s, h_s, A1, t, 2);
        wmma_gemm_k<<<32, 128, 0, stream>>>(A1, WpN1, gates, KPN);
        lstm_cell_k<<<(B_ * H_ + 255) / 256, 256, 0, stream>>>(
            gates, b_ihn + G4H, b_hhn + G4H, h_s + (size_t)2 * B_ * H_, c_s + (size_t)2 * B_ * H_);

        // output projection
        out_k<<<B_, 128, 0, stream>>>(h_s, W_out, b_out, out_main, t);
    }

    // ---- final states ----
    (void)hipMemcpyAsync(out_h, h_s, sizeof(float) * L_ * B_ * H_, hipMemcpyDeviceToDevice, stream);
    (void)hipMemcpyAsync(out_c, c_s, sizeof(float) * L_ * B_ * H_, hipMemcpyDeviceToDevice, stream);
    (void)hipMemcpyAsync(out_w, w_s, sizeof(float) * B_ * ENC_, hipMemcpyDeviceToDevice, stream);
    (void)hipMemcpyAsync(out_kf, k_s, sizeof(float) * B_ * KATT, hipMemcpyDeviceToDevice, stream);
}
